// Florence2VisionWindowAttention_62612033241730
// MI455X (gfx1250) — compile-verified, matrix operand order checked by reading, not run on hardware
//
#include <hip/hip_runtime.h>
#include <hip/hip_bf16.h>

typedef __attribute__((ext_vector_type(16))) __bf16 v16bf;
typedef __attribute__((ext_vector_type(8)))  __bf16 v8bf;
typedef __attribute__((ext_vector_type(8)))  float  v8f;
typedef __attribute__((ext_vector_type(4)))  int    v4i;

#define NT    144      // tokens per 12x12 window
#define NH    16
#define HD    32
#define CDIM  512
#define NWIN  128      // 8 batches * 4 * 4 windows
#define QKVN  1536

#if __has_builtin(__builtin_amdgcn_global_load_async_to_lds_b128) && \
    __has_builtin(__builtin_amdgcn_s_wait_asynccnt)
#define USE_ASYNC_LDS 1
typedef __attribute__((address_space(1))) v4i* async_gptr;
typedef __attribute__((address_space(3))) v4i* async_lptr;
#else
#define USE_ASYNC_LDS 0
#endif

__device__ __forceinline__ __bf16 f2bf(float f) { return (__bf16)f; }

// Load one 16x32 bf16 WMMA operand fragment for this lane.
// Layout (CDNA5 ISA 7.12.2, 16-bit A / B): lanes 0-15 and 16-31 both map to
// rows/cols 0-15; lane>=16 takes K-base 8.  Per lane: K {base..base+7} in
// VGPR0-3 and K {base+16..base+23} in VGPR4-7 -> two contiguous 16B loads.
__device__ __forceinline__ v16bf load_frag(const __bf16* rowp, int lane) {
  const int base = (lane & 16) ? 8 : 0;
  v8bf lo = *(const v8bf*)(rowp + base);
  v8bf hi = *(const v8bf*)(rowp + base + 16);
  v16bf r;
#pragma unroll
  for (int i = 0; i < 8; ++i) { r[i] = lo[i]; r[i + 8] = hi[i]; }
  return r;
}

// 9-tile WMMA column sweep.  sched_group_barrier enforces
// (4 ds-reads)(wmma)[(2 ds-reads)(wmma)]x8 so each WMMA's A-fragment loads
// were issued one group ahead -> s_wait_dscnt can tolerate in-flight loads.
__device__ __forceinline__ void wmma_col9(const __bf16* A, const v16bf bfrag,
                                          v8f acc[9], int lane) {
  v16bf af = load_frag(&A[(lane & 15) * 32], lane);
#pragma unroll
  for (int mt = 0; mt < 9; ++mt) {
    v16bf afn;
    if (mt < 8) afn = load_frag(&A[((mt + 1) * 16 + (lane & 15)) * 32], lane);
    acc[mt] = __builtin_amdgcn_wmma_f32_16x16x32_bf16(
        false, af, false, bfrag, (short)0, acc[mt], false, false);
    af = afn;
  }
#if __has_builtin(__builtin_amdgcn_sched_group_barrier)
  __builtin_amdgcn_sched_group_barrier(0x100, 4, 0);   // bfrag + afrag0
#pragma unroll
  for (int g = 0; g < 8; ++g) {
    __builtin_amdgcn_sched_group_barrier(0x008, 1, 0); // WMMA g
    __builtin_amdgcn_sched_group_barrier(0x100, 2, 0); // afrag g+1
  }
  __builtin_amdgcn_sched_group_barrier(0x008, 1, 0);   // WMMA 8
#endif
}

// ---------------------------------------------------------------------------
// Kernel 1: QKV GEMM.  grid = (128 windows, 12 N-slices), 256 threads.
// Double-buffered LDS; per-thread staging addresses hoisted out of the K loop.
// ---------------------------------------------------------------------------
__global__ void __launch_bounds__(256)
qkv_gemm_kernel(const float* __restrict__ x, const float* __restrict__ qkv_w,
                const float* __restrict__ qkv_b, __bf16* __restrict__ q_ws,
                __bf16* __restrict__ k_ws, __bf16* __restrict__ v_ws) {
  __shared__ __bf16 Alds[2][NT * 32];    // 2 x 144x32
  __shared__ __bf16 Blds[2][128 * 32];   // 2 x N-major [n][k]

  const int w    = blockIdx.x;
  const int nsl  = blockIdx.y;
  const int tid  = threadIdx.x;
  const int lane = tid & 31;
  const int wv   = tid >> 5;

  const int b  = w >> 4;
  const int wh = (w >> 2) & 3;
  const int ww = w & 3;
  const long winbase = (((long)b * 48 + wh * 12) * 48 + ww * 12) * CDIM;
  const int  nbase   = nsl * 128;

  // Hoisted per-thread staging addresses (A: 4(+1) granules, B: 4 granules)
  const float* gA[4]; int lA[4];
#pragma unroll
  for (int i = 0; i < 4; ++i) {
    const int e = tid + i * 256;               // < 1024
    const int t = e >> 3, q4 = e & 7;
    const int r = t / 12, cj = t % 12;
    gA[i] = x + winbase + (long)(r * 48 + cj) * CDIM + q4 * 4;
    lA[i] = t * 32 + q4 * 4;
  }
  const float* gA4 = x; int lA4 = 0;
  if (tid < 128) {
    const int e = tid + 1024;                  // 1024..1151
    const int t = e >> 3, q4 = e & 7;
    const int r = t / 12, cj = t % 12;
    gA4 = x + winbase + (long)(r * 48 + cj) * CDIM + q4 * 4;
    lA4 = t * 32 + q4 * 4;
  }
  const float* gB[4]; int lB[4];
#pragma unroll
  for (int i = 0; i < 4; ++i) {
    const int e = tid + i * 256;               // < 1024
    const int k = e >> 5, nq = e & 31;
    gB[i] = qkv_w + (long)k * QKVN + nbase + nq * 4;
    lB[i] = (nq * 4) * 32 + k;
  }

  v8f acc[9];
#pragma unroll
  for (int m = 0; m < 9; ++m) acc[m] = v8f{};

  auto stage_a4 = [&](const float* gp, __bf16* lp) {
    const float4 f = *(const float4*)gp;
    union { __bf16 h[4]; unsigned int u[2]; } pk;
    pk.h[0] = f2bf(f.x); pk.h[1] = f2bf(f.y);
    pk.h[2] = f2bf(f.z); pk.h[3] = f2bf(f.w);
    *(unsigned int*)(lp + 0) = pk.u[0];
    *(unsigned int*)(lp + 2) = pk.u[1];
  };

  auto load_tiles = [&](int ks, int buf) {
    const int c0 = ks * 32;
#pragma unroll
    for (int i = 0; i < 4; ++i) stage_a4(gA[i] + c0, &Alds[buf][lA[i]]);
    if (tid < 128) stage_a4(gA4 + c0, &Alds[buf][lA4]);
#pragma unroll
    for (int i = 0; i < 4; ++i) {
      const float4 f = *(const float4*)(gB[i] + (long)c0 * QKVN);
      __bf16* lp = &Blds[buf][lB[i]];
      lp[0]  = f2bf(f.x);
      lp[32] = f2bf(f.y);
      lp[64] = f2bf(f.z);
      lp[96] = f2bf(f.w);
    }
  };

  load_tiles(0, 0);
  for (int ks = 0; ks < 16; ++ks) {
    const int buf = ks & 1;
    __syncthreads();                       // tile ks visible; buf^1 free
    if (ks < 15) load_tiles(ks + 1, buf ^ 1);
    const v16bf bfrag =
        load_frag(&Blds[buf][(wv * 16 + (lane & 15)) * 32], lane);
    wmma_col9(Alds[buf], bfrag, acc, lane);
  }

  // Epilogue: + bias, scatter to q/k/v bf16 regions
  const int ncol  = nbase + wv * 16 + (lane & 15);
  const float bias = qkv_b[ncol];
  const int which = ncol >> 9;
  const int head  = (ncol >> 5) & 15;
  const int d     = ncol & 31;
  __bf16* dst = (which == 0) ? q_ws : (which == 1) ? k_ws : v_ws;
  const long obase = (long)(w * NH + head) * NT * HD + d;
  const int  rofs  = (lane & 16) ? 8 : 0;
#pragma unroll
  for (int mt = 0; mt < 9; ++mt)
#pragma unroll
    for (int vr = 0; vr < 8; ++vr) {
      const int row = mt * 16 + vr + rofs;
      dst[obase + (long)row * HD] = f2bf(acc[mt][vr] + bias);
    }
}

// ---------------------------------------------------------------------------
// Kernel 2: per-(window, head) attention. grid = 2048, 256 threads.
// ---------------------------------------------------------------------------
__global__ void __launch_bounds__(256)
attn_kernel(const __bf16* __restrict__ q_ws, const __bf16* __restrict__ k_ws,
            const __bf16* __restrict__ v_ws, __bf16* __restrict__ ao_ws) {
  __shared__ __bf16 vT[32 * 160];     // v transposed [dim][token], zero-padded
  __shared__ __bf16 at[NT * 160];     // scores, then softmaxed attn (bf16)

  const int w    = blockIdx.x >> 4;
  const int h    = blockIdx.x & 15;
  const int tid  = threadIdx.x;
  const int lane = tid & 31;
  const int wv   = tid >> 5;
  const long base = (long)(w * NH + h) * NT * HD;

  // Stage v^T into LDS, zero-pad tokens [144,160)
  for (int e = tid; e < NT * HD; e += 256) {
    const int t = e >> 5, d = e & 31;
    vT[d * 160 + t] = v_ws[base + e];
  }
  for (int e = tid; e < 32 * 16; e += 256) {
    const int d = e >> 4, t = 144 + (e & 15);
    vT[d * 160 + t] = (__bf16)0.0f;
  }
  __syncthreads();

  const float scale = 0.17677669529663687f;  // 1/sqrt(32)

  // scores = q @ k^T : 81 tiles, single K=32 WMMA each
  for (int idx = wv; idx < 81; idx += 8) {
    const int mt = idx / 9, nt = idx % 9;
    const v16bf afrag = load_frag(q_ws + base + (mt * 16 + (lane & 15)) * 32, lane);
    const v16bf bfrag = load_frag(k_ws + base + (nt * 16 + (lane & 15)) * 32, lane);
    v8f acc = v8f{};
    acc = __builtin_amdgcn_wmma_f32_16x16x32_bf16(
        false, afrag, false, bfrag, (short)0, acc, false, false);
    const int rofs = (lane & 16) ? 8 : 0;
    const int col  = nt * 16 + (lane & 15);
#pragma unroll
    for (int vr = 0; vr < 8; ++vr)
      at[(mt * 16 + vr + rofs) * 160 + col] = f2bf(acc[vr] * scale);
  }
  __syncthreads();

  // Row softmax (threads 0..143), 3 passes over LDS, no private spill
  if (tid < NT) {
    __bf16* row = at + tid * 160;
    float m = -1e30f;
    for (int c = 0; c < NT; ++c) m = fmaxf(m, (float)row[c]);
    float s = 0.0f;
    for (int c = 0; c < NT; ++c) s += __expf((float)row[c] - m);
    const float inv = 1.0f / s;
    for (int c = 0; c < NT; ++c) row[c] = f2bf(__expf((float)row[c] - m) * inv);
    for (int c = NT; c < 160; ++c) row[c] = (__bf16)0.0f;
  }
  __syncthreads();

  // out = attn @ v : 9x2 tiles, K = 160 -> 5 WMMA steps
  for (int idx = wv; idx < 18; idx += 8) {
    const int mt = idx >> 1, nt = idx & 1;
    v8f acc = v8f{};
#pragma unroll
    for (int ksv = 0; ksv < 5; ++ksv) {
      const v16bf afrag = load_frag(&at[(mt * 16 + (lane & 15)) * 160 + ksv * 32], lane);
      const v16bf bfrag = load_frag(&vT[(nt * 16 + (lane & 15)) * 160 + ksv * 32], lane);
      acc = __builtin_amdgcn_wmma_f32_16x16x32_bf16(
          false, afrag, false, bfrag, (short)0, acc, false, false);
    }
    const int rofs = (lane & 16) ? 8 : 0;
    const int d    = nt * 16 + (lane & 15);
#pragma unroll
    for (int vr = 0; vr < 8; ++vr) {
      const int row = mt * 16 + vr + rofs;
      ao_ws[(long)(w * NT + row) * CDIM + h * HD + d] = f2bf(acc[vr]);
    }
  }
}

// ---------------------------------------------------------------------------
// Kernel 3: proj GEMM + bias + window-reverse scatter to fp32 output.
// A tile is a pure bf16 copy -> GLOBAL_LOAD_ASYNC_TO_LDS_B128 (ASYNCcnt path).
// ---------------------------------------------------------------------------
__global__ void __launch_bounds__(256)
proj_gemm_kernel(const __bf16* __restrict__ ao_ws, const float* __restrict__ proj_w,
                 const float* __restrict__ proj_b, float* __restrict__ out) {
  __shared__ __bf16 Alds[2][NT * 32];
  __shared__ __bf16 Blds[2][128 * 32];

  const int w    = blockIdx.x;
  const int nsl  = blockIdx.y;
  const int tid  = threadIdx.x;
  const int lane = tid & 31;
  const int wv   = tid >> 5;
  const int nbase = nsl * 128;

  // Hoisted staging addresses: A = 2(+1) 16B granules, B = 4 granules
  const __bf16* gA[2]; int lA[2];
#pragma unroll
  for (int i = 0; i < 2; ++i) {
    const int e = tid + i * 256;               // < 512
    const int t = e >> 2, q8 = e & 3;
    gA[i] = ao_ws + (long)(w * NT + t) * CDIM + q8 * 8;
    lA[i] = t * 32 + q8 * 8;
  }
  const __bf16* gA2 = ao_ws; int lA2 = 0;
  if (tid < 64) {
    const int e = tid + 512;                   // 512..575
    const int t = e >> 2, q8 = e & 3;
    gA2 = ao_ws + (long)(w * NT + t) * CDIM + q8 * 8;
    lA2 = t * 32 + q8 * 8;
  }
  const float* gB[4]; int lB[4];
#pragma unroll
  for (int i = 0; i < 4; ++i) {
    const int e = tid + i * 256;               // < 1024
    const int k = e >> 5, nq = e & 31;
    gB[i] = proj_w + (long)k * CDIM + nbase + nq * 4;
    lB[i] = (nq * 4) * 32 + k;
  }

  v8f acc[9];
#pragma unroll
  for (int m = 0; m < 9; ++m) acc[m] = v8f{};

  auto copy_a = [&](const __bf16* gp, __bf16* lp) {
#if USE_ASYNC_LDS
    __builtin_amdgcn_global_load_async_to_lds_b128(
        (async_gptr)gp, (async_lptr)lp, 0, 0);
#else
    *(uint4*)lp = *(const uint4*)gp;
#endif
  };

  auto load_tiles = [&](int ks, int buf) {
    const int c0 = ks * 32;
#pragma unroll
    for (int i = 0; i < 2; ++i) copy_a(gA[i] + c0, &Alds[buf][lA[i]]);
    if (tid < 64) copy_a(gA2 + c0, &Alds[buf][lA2]);
#pragma unroll
    for (int i = 0; i < 4; ++i) {
      const float4 f = *(const float4*)(gB[i] + (long)c0 * CDIM);
      __bf16* lp = &Blds[buf][lB[i]];
      lp[0]  = f2bf(f.x);
      lp[32] = f2bf(f.y);
      lp[64] = f2bf(f.z);
      lp[96] = f2bf(f.w);
    }
  };

  load_tiles(0, 0);
  for (int ks = 0; ks < 16; ++ks) {
    const int buf = ks & 1;
#if USE_ASYNC_LDS
    __builtin_amdgcn_s_wait_asynccnt(0);   // async fills for tile ks done
#endif
    __syncthreads();
    if (ks < 15) load_tiles(ks + 1, buf ^ 1);
    const v16bf bfrag =
        load_frag(&Blds[buf][(wv * 16 + (lane & 15)) * 32], lane);
    wmma_col9(Alds[buf], bfrag, acc, lane);
  }

  // Epilogue: + bias, window_reverse scatter into (8, 2304, 512) fp32
  const int ncol  = nbase + wv * 16 + (lane & 15);
  const float bias = proj_b[ncol];
  const int b  = w >> 4;
  const int wh = (w >> 2) & 3;
  const int ww = w & 3;
  const int rofs = (lane & 16) ? 8 : 0;
#pragma unroll
  for (int mt = 0; mt < 9; ++mt)
#pragma unroll
    for (int vr = 0; vr < 8; ++vr) {
      const int t = mt * 16 + vr + rofs;
      const int i = t / 12, j = t % 12;
      const long oidx =
          ((long)b * 2304 + (wh * 12 + i) * 48 + (ww * 12 + j)) * CDIM + ncol;
      out[oidx] = acc[mt][vr] + bias;
    }
}

// ---------------------------------------------------------------------------
extern "C" void kernel_launch(void* const* d_in, const int* in_sizes, int n_in,
                              void* d_out, int out_size, void* d_ws, size_t ws_size,
                              hipStream_t stream) {
  const float* x      = (const float*)d_in[0];
  const float* qkv_w  = (const float*)d_in[1];
  const float* qkv_b  = (const float*)d_in[2];
  const float* proj_w = (const float*)d_in[3];
  const float* proj_b = (const float*)d_in[4];
  float* out = (float*)d_out;

  // Workspace: q, k, v regions [win][head][tok][dim] + attn-out [win][tok][c]
  const long NQ = (long)NWIN * NH * NT * HD;   // 9,437,184 bf16 each
  __bf16* q_ws  = (__bf16*)d_ws;
  __bf16* k_ws  = q_ws + NQ;
  __bf16* v_ws  = k_ws + NQ;
  __bf16* ao_ws = v_ws + NQ;                   // 9,437,184 bf16

  dim3 g1(NWIN, 12);
  qkv_gemm_kernel<<<g1, 256, 0, stream>>>(x, qkv_w, qkv_b, q_ws, k_ws, v_ws);

  attn_kernel<<<dim3(NWIN * NH), 256, 0, stream>>>(q_ws, k_ws, v_ws, ao_ws);

  dim3 g3(NWIN, 4);
  proj_gemm_kernel<<<g3, 256, 0, stream>>>(ao_ws, proj_w, proj_b, out);
}
